// AgentModel_6262062317903
// MI455X (gfx1250) — matrix operand order repaired and checked by reading, not compiled, not run on hardware
//
#include <hip/hip_runtime.h>
#include <hip/hip_bf16.h>

typedef _Float16 h16 __attribute__((ext_vector_type(16)));
typedef _Float16 h8  __attribute__((ext_vector_type(8)));
typedef float    f8  __attribute__((ext_vector_type(8)));

#define N_NODES 131072
#define N_EDGES 1048576
#define NB      32
#define HC      128    // H*C
#define CC      64     // per-head channels
#define KV      262144 // 64*64*64

// ---------------- workspace layout (float offsets) ----------------
#define WS_BIG   ((size_t)0)            // 16,777,216 floats: conv v1/v2/v3, then h (per layer)
#define WS_V1    (WS_BIG)               // 2,097,152
#define WS_V2    (WS_BIG + 2097152)     // 4,194,304
#define WS_V3    (WS_BIG + 6291456)     // 8,388,608
#define WS_OUT   ((size_t)16777216)     // 16,777,216 floats: GAT out accumulator / layer input
#define WS_SD    ((size_t)33554432)     // 524,288: s at +0, d at +262144
#define WS_M     ((size_t)34078720)     // 262,144
#define WS_DEN   ((size_t)34340864)     // 262,144
#define WS_EB    ((size_t)34603008)     // 2,097,152 (E*H): e then p
#define WS_VF    ((size_t)36700160)     // 4096 vision feature [32,128]
#define WS_GP    ((size_t)36704256)     // 4096 pooled graph  [32,128]
#define WS_HC    ((size_t)36708352)     // 4096 head hidden   [32,128]

__device__ __forceinline__ void atomicMaxF(float* addr, float val) {
  if (val >= 0.0f) atomicMax((int*)addr, __float_as_int(val));
  else             atomicMin((unsigned int*)addr, __float_as_uint(val));
}

// ---------------- vision: direct 3x3 SAME conv + relu ----------------
__global__ void k_conv3x3_relu(const float* __restrict__ x, const float* __restrict__ w,
                               const float* __restrict__ bias, float* __restrict__ y,
                               int Cin, int Cout) {
  int tid = blockIdx.x * blockDim.x + threadIdx.x;
  int total = NB * Cout * 64 * 64;
  if (tid >= total) return;
  int px = tid & 63, py = (tid >> 6) & 63;
  int co = (tid >> 12) % Cout, b = tid / (Cout << 12);
  float acc = bias[co];
  for (int ci = 0; ci < Cin; ++ci) {
    const float* xp = x + ((size_t)(b * Cin + ci)) * 4096;
    const float* wp = w + ((size_t)(co * Cin + ci)) * 9;
    #pragma unroll
    for (int ky = -1; ky <= 1; ++ky) {
      int yy = py + ky;
      if ((unsigned)yy >= 64u) continue;
      #pragma unroll
      for (int kx = -1; kx <= 1; ++kx) {
        int xx = px + kx;
        if ((unsigned)xx >= 64u) continue;
        acc += xp[yy * 64 + xx] * wp[(ky + 1) * 3 + (kx + 1)];
      }
    }
  }
  y[tid] = fmaxf(acc, 0.0f);
}

// ---------------- WMMA GEMM via LDS staging: C[M,128] = A[M,K] @ W[K,128] ----------------
// block = 256 threads (8 waves); each block computes a 16x128 strip (wave w -> n-tile w).
// A (16xK) and W (Kx128, transposed per-column) staged in LDS as f16 so that WMMA
// fragments are contiguous ds_load_b128 reads. K templated (16 or 128), padded to KP.
template <int K>
__global__ void k_gemm_wmma(const float* __restrict__ A, const float* __restrict__ Wm,
                            float* __restrict__ C) {
  constexpr int KP = (K < 32) ? 32 : K;
  __shared__ __attribute__((aligned(32))) _Float16 aLds[16 * KP];
  __shared__ __attribute__((aligned(32))) _Float16 wLds[128 * KP];   // [n][KP]
  const int t = threadIdx.x;
  const int m0 = blockIdx.x * 16;

  // stage A strip (16 rows x K): contiguous float4 global loads, f16 into LDS
  const float* Ab = A + (size_t)m0 * K;
  for (int idx = t; idx < (16 * K) / 4; idx += 256) {
    int fl = idx * 4;
    int r = fl / K, c = fl % K;
    const float4 v = *(const float4*)(Ab + fl);
    _Float16* dst = aLds + r * KP + c;
    dst[0] = (_Float16)v.x; dst[1] = (_Float16)v.y;
    dst[2] = (_Float16)v.z; dst[3] = (_Float16)v.w;
  }
  // stage W (Kx128) transposed -> wLds[n][k]
  for (int idx = t; idx < (K * 128) / 4; idx += 256) {
    int fl = idx * 4;
    int k = fl >> 7, n = fl & 127;
    const float4 v = *(const float4*)(Wm + (size_t)k * 128 + n);
    wLds[(n + 0) * KP + k] = (_Float16)v.x;
    wLds[(n + 1) * KP + k] = (_Float16)v.y;
    wLds[(n + 2) * KP + k] = (_Float16)v.z;
    wLds[(n + 3) * KP + k] = (_Float16)v.w;
  }
  if constexpr (K < KP) {   // zero the K..KP-1 pad columns (disjoint from staged region)
    for (int idx = t; idx < 16 * (KP - K); idx += 256) {
      int r = idx / (KP - K), c = K + idx % (KP - K);
      aLds[r * KP + c] = (_Float16)0.f;
    }
    for (int idx = t; idx < 128 * (KP - K); idx += 256) {
      int r = idx / (KP - K), c = K + idx % (KP - K);
      wLds[r * KP + c] = (_Float16)0.f;
    }
  }
  __syncthreads();

  const int lane = t & 31, wv = t >> 5;
  const int lrow = lane & 15, hi = lane >> 4;
  const int n0 = wv * 16;
  f8 acc = {0.f, 0.f, 0.f, 0.f, 0.f, 0.f, 0.f, 0.f};
  const _Float16* arow = aLds + lrow * KP;
  const _Float16* bcol = wLds + (n0 + lrow) * KP;
  #pragma unroll
  for (int k0 = 0; k0 < KP; k0 += 32) {
    // A fragment: halves 0-15 <- K = k0+16*hi+(0..15) contiguous
    h8 r1 = *(const h8*)(arow + k0 + 16 * hi);
    h8 r2 = *(const h8*)(arow + k0 + 16 * hi + 8);
    // B fragment: col n0+lrow, K = k0+16*hi+(0..15) contiguous
    h16 b = *(const h16*)(bcol + k0 + 16 * hi);
    h16 a;
    #pragma unroll
    for (int h = 0; h < 8; ++h) { a[h] = r1[h]; a[h + 8] = r2[h]; }
    acc = __builtin_amdgcn_wmma_f32_16x16x32_f16(false, a, false, b, (short)0, acc, false, false);
  }
  const int row = m0 + 8 * hi;
  #pragma unroll
  for (int r = 0; r < 8; ++r)
    C[(size_t)(row + r) * 128 + (n0 + lrow)] = acc[r];
}

// ---------------- fc_v: [32,262144]@[262144,128], split-K WMMA + atomic reduce ----------------
// grid = (2, 8, KSPLIT), block = 32; C preinitialized with bias. No predicated loads.
__global__ void k_fcv_wmma(const float* __restrict__ A, const float* __restrict__ Wm,
                           float* __restrict__ C) {
  int m0 = blockIdx.x * 16, n0 = blockIdx.y * 16;
  int lane = threadIdx.x, lrow = lane & 15, hi = lane >> 4;
  int KC = KV / gridDim.z;
  int kbeg = blockIdx.z * KC, kend = kbeg + KC;
  f8 acc = {0.f, 0.f, 0.f, 0.f, 0.f, 0.f, 0.f, 0.f};
  const float* arow = A + (size_t)(m0 + lrow) * KV;
  for (int k0 = kbeg; k0 < kend; k0 += 32) {
    __builtin_prefetch(&Wm[(size_t)(k0 + 64) * HC + n0 + lrow], 0, 3);
    // A fragment: one contiguous 16-float run at k0+16*hi
    const float4 q0 = *(const float4*)(arow + k0 + 16 * hi);
    const float4 q1 = *(const float4*)(arow + k0 + 16 * hi + 4);
    const float4 q2 = *(const float4*)(arow + k0 + 16 * hi + 8);
    const float4 q3 = *(const float4*)(arow + k0 + 16 * hi + 12);
    h16 a;
    a[0]  = (_Float16)q0.x; a[1]  = (_Float16)q0.y; a[2]  = (_Float16)q0.z; a[3]  = (_Float16)q0.w;
    a[4]  = (_Float16)q1.x; a[5]  = (_Float16)q1.y; a[6]  = (_Float16)q1.z; a[7]  = (_Float16)q1.w;
    a[8]  = (_Float16)q2.x; a[9]  = (_Float16)q2.y; a[10] = (_Float16)q2.z; a[11] = (_Float16)q2.w;
    a[12] = (_Float16)q3.x; a[13] = (_Float16)q3.y; a[14] = (_Float16)q3.z; a[15] = (_Float16)q3.w;
    // B fragment: lanes 0-15 read 16 consecutive floats of row k0+h; lanes 16-31 of row k0+h+16
    h16 b;
    #pragma unroll
    for (int h = 0; h < 16; ++h)
      b[h] = (_Float16)Wm[(size_t)(k0 + h + hi * 16) * HC + n0 + lrow];
    acc = __builtin_amdgcn_wmma_f32_16x16x32_f16(false, a, false, b, (short)0, acc, false, false);
  }
  int row = m0 + 8 * hi;
  #pragma unroll
  for (int r = 0; r < 8; ++r)
    atomicAdd(&C[(size_t)(row + r) * HC + (n0 + lrow)], acc[r]);
}

// ---------------- small helpers ----------------
__global__ void k_fill_bias_rows(float* __restrict__ dst, const float* __restrict__ bias,
                                 int rows, int cols) {
  size_t tid = (size_t)blockIdx.x * blockDim.x + threadIdx.x;
  if (tid >= (size_t)rows * cols) return;
  dst[tid] = bias[tid % cols];
}

__global__ void k_fill_neginf(float* __restrict__ dst, int n) {
  int tid = blockIdx.x * blockDim.x + threadIdx.x;
  if (tid < n) dst[tid] = -__builtin_inff();
}

// s/d attention projections: tid over N_NODES*2 (node, head)
__global__ void k_sd(const float* __restrict__ hbuf, const float* __restrict__ a_src,
                     const float* __restrict__ a_dst, float* __restrict__ sd) {
  int tid = blockIdx.x * blockDim.x + threadIdx.x;
  if (tid >= N_NODES * 2) return;
  int node = tid >> 1, hh = tid & 1;
  const float* hv = hbuf + (size_t)node * HC + hh * CC;
  const float* as = a_src + hh * CC;
  const float* ad = a_dst + hh * CC;
  float s = 0.f, d = 0.f;
  #pragma unroll 8
  for (int c = 0; c < CC; ++c) { s += hv[c] * as[c]; d += hv[c] * ad[c]; }
  sd[tid] = s;
  sd[262144 + tid] = d;
}

// e = leaky_relu(s[src]+d[dst]) with mask; segment-max into mbuf
__global__ void k_edge_e(const int* __restrict__ eidx, const float* __restrict__ eattr,
                         const float* __restrict__ sd, float* __restrict__ ebuf,
                         float* __restrict__ mbuf) {
  int tid = blockIdx.x * blockDim.x + threadIdx.x;
  if (tid >= N_EDGES * 2) return;
  int e = tid >> 1, hh = tid & 1;
  int src = eidx[e], dst = eidx[N_EDGES + e];
  float val;
  if (eattr[(size_t)e * 4 + 3] == 1.0f) {
    float v = sd[src * 2 + hh] + sd[262144 + dst * 2 + hh];
    val = (v < 0.f) ? 0.2f * v : v;
  } else {
    val = -__builtin_inff();
  }
  ebuf[tid] = val;
  atomicMaxF(&mbuf[dst * 2 + hh], val);
}

// m = finite(m)?m:0 ; denom = 0
__global__ void k_fix_m(float* __restrict__ mbuf, float* __restrict__ den, int n) {
  int tid = blockIdx.x * blockDim.x + threadIdx.x;
  if (tid >= n) return;
  float m = mbuf[tid];
  if (m == -__builtin_inff()) mbuf[tid] = 0.f;
  den[tid] = 0.f;
}

// p = exp(e - m[dst]) (0 if masked); segment-sum into denom
__global__ void k_edge_p(const int* __restrict__ eidx, float* __restrict__ ebuf,
                         const float* __restrict__ mbuf, float* __restrict__ den) {
  int tid = blockIdx.x * blockDim.x + threadIdx.x;
  if (tid >= N_EDGES * 2) return;
  int e = tid >> 1, hh = tid & 1;
  int dst = eidx[N_EDGES + e];
  float val = ebuf[tid];
  float p = (val > -3.0e38f) ? __expf(val - mbuf[dst * 2 + hh]) : 0.f;
  ebuf[tid] = p;
  atomicAdd(&den[dst * 2 + hh], p);
}

// out[dst,c] += h[src,c] * p/(den[dst]+eps) ; tid over E*128
__global__ void k_edge_aggr(const int* __restrict__ eidx, const float* __restrict__ ebuf,
                            const float* __restrict__ den, const float* __restrict__ hbuf,
                            float* __restrict__ outb) {
  size_t tid = (size_t)blockIdx.x * blockDim.x + threadIdx.x;
  if (tid >= (size_t)N_EDGES * HC) return;
  int e = (int)(tid >> 7), c = (int)(tid & 127), hh = c >> 6;
  int src = eidx[e], dst = eidx[N_EDGES + e];
  float alpha = ebuf[e * 2 + hh] / (den[dst * 2 + hh] + 1e-16f);
  atomicAdd(&outb[(size_t)dst * HC + c], hbuf[(size_t)src * HC + c] * alpha);
}

__global__ void k_relu(float* __restrict__ x, size_t n) {
  size_t tid = (size_t)blockIdx.x * blockDim.x + threadIdx.x;
  if (tid < n) x[tid] = fmaxf(x[tid], 0.f);
}

// mean pool per graph: contiguous 4096 nodes per graph; block=(128), grid=(32)
__global__ void k_pool(const float* __restrict__ x, float* __restrict__ g) {
  int b = blockIdx.x, c = threadIdx.x;
  const float* xp = x + (size_t)b * 4096 * HC + c;
  float acc = 0.f;
  for (int i = 0; i < 4096; ++i) acc += xp[(size_t)i * HC];
  g[b * HC + c] = acc * (1.0f / 4096.0f);
}

// head fc1: [32,256]@[256,128]+b, relu (concat read from vfeat/gpool)
__global__ void k_fc1(const float* __restrict__ vf, const float* __restrict__ gp,
                      const float* __restrict__ w, const float* __restrict__ bias,
                      float* __restrict__ out) {
  int tid = blockIdx.x * blockDim.x + threadIdx.x;
  if (tid >= NB * HC) return;
  int m = tid >> 7, n = tid & 127;
  float acc = bias[n];
  #pragma unroll 8
  for (int k = 0; k < 128; ++k) acc += vf[m * 128 + k] * w[k * 128 + n];
  #pragma unroll 8
  for (int k = 0; k < 128; ++k) acc += gp[m * 128 + k] * w[(128 + k) * 128 + n];
  out[tid] = fmaxf(acc, 0.f);
}

// head fc2: [32,128]@[128,8]+b -> d_out
__global__ void k_fc2(const float* __restrict__ hc, const float* __restrict__ w,
                      const float* __restrict__ bias, float* __restrict__ out) {
  int tid = blockIdx.x * blockDim.x + threadIdx.x;
  if (tid >= NB * 8) return;
  int m = tid >> 3, n = tid & 7;
  float acc = bias[n];
  #pragma unroll 8
  for (int k = 0; k < 128; ++k) acc += hc[m * 128 + k] * w[k * 8 + n];
  out[tid] = acc;
}

extern "C" void kernel_launch(void* const* d_in, const int* in_sizes, int n_in,
                              void* d_out, int out_size, void* d_ws, size_t ws_size,
                              hipStream_t stream) {
  const float* vision = (const float*)d_in[0];
  const float* nodef  = (const float*)d_in[1];
  const float* eattr  = (const float*)d_in[2];
  const int*   eidx   = (const int*)d_in[3];
  const float* w_c1 = (const float*)d_in[5],  *b_c1 = (const float*)d_in[6];
  const float* w_c2 = (const float*)d_in[7],  *b_c2 = (const float*)d_in[8];
  const float* w_c3 = (const float*)d_in[9],  *b_c3 = (const float*)d_in[10];
  const float* w_fcv = (const float*)d_in[11], *b_fcv = (const float*)d_in[12];
  const float* W1 = (const float*)d_in[13];
  const float* a_src1 = (const float*)d_in[14], *a_dst1 = (const float*)d_in[15];
  const float* b1 = (const float*)d_in[16];
  const float* W2 = (const float*)d_in[17];
  const float* a_src2 = (const float*)d_in[18], *a_dst2 = (const float*)d_in[19];
  const float* b2 = (const float*)d_in[20];
  const float* w_fc1 = (const float*)d_in[21], *b_fc1 = (const float*)d_in[22];
  const float* w_fc2 = (const float*)d_in[23], *b_fc2 = (const float*)d_in[24];

  float* ws = (float*)d_ws;
  float* big  = ws + WS_BIG;   // h matrix region (also conv scratch)
  float* outb = ws + WS_OUT;   // GAT accumulator / layer input
  float* sd   = ws + WS_SD;
  float* mbuf = ws + WS_M;
  float* den  = ws + WS_DEN;
  float* ebuf = ws + WS_EB;
  float* vf   = ws + WS_VF;
  float* gp   = ws + WS_GP;
  float* hc   = ws + WS_HC;

  const int T = 256;

  // ---- vision branch (uses BIG region as conv scratch; runs before graph) ----
  k_conv3x3_relu<<<(NB*16*4096 + T-1)/T, T, 0, stream>>>(vision, w_c1, b_c1, ws + WS_V1, 3, 16);
  k_conv3x3_relu<<<(NB*32*4096 + T-1)/T, T, 0, stream>>>(ws + WS_V1, w_c2, b_c2, ws + WS_V2, 16, 32);
  k_conv3x3_relu<<<(NB*64*4096 + T-1)/T, T, 0, stream>>>(ws + WS_V2, w_c3, b_c3, ws + WS_V3, 32, 64);
  k_fill_bias_rows<<<(NB*HC + T-1)/T, T, 0, stream>>>(vf, b_fcv, NB, HC);
  k_fcv_wmma<<<dim3(2, 8, 512), 32, 0, stream>>>(ws + WS_V3, w_fcv, vf);

  // ---- GAT layer 1 ----
  k_gemm_wmma<16><<<N_NODES/16, 256, 0, stream>>>(nodef, W1, big);
  k_sd<<<(N_NODES*2 + T-1)/T, T, 0, stream>>>(big, a_src1, a_dst1, sd);
  k_fill_neginf<<<(N_NODES*2 + T-1)/T, T, 0, stream>>>(mbuf, N_NODES*2);
  k_edge_e<<<(N_EDGES*2 + T-1)/T, T, 0, stream>>>(eidx, eattr, sd, ebuf, mbuf);
  k_fix_m<<<(N_NODES*2 + T-1)/T, T, 0, stream>>>(mbuf, den, N_NODES*2);
  k_edge_p<<<(N_EDGES*2 + T-1)/T, T, 0, stream>>>(eidx, ebuf, mbuf, den);
  k_fill_bias_rows<<<(N_NODES*HC + T-1)/T, T, 0, stream>>>(outb, b1, N_NODES, HC);
  k_edge_aggr<<<(int)(((size_t)N_EDGES*HC + T-1)/T), T, 0, stream>>>(eidx, ebuf, den, big, outb);
  k_relu<<<(N_NODES*HC + T-1)/T, T, 0, stream>>>(outb, (size_t)N_NODES*HC);

  // ---- GAT layer 2 (input = outb, h -> big, accumulate back into outb) ----
  k_gemm_wmma<128><<<N_NODES/16, 256, 0, stream>>>(outb, W2, big);
  k_sd<<<(N_NODES*2 + T-1)/T, T, 0, stream>>>(big, a_src2, a_dst2, sd);
  k_fill_neginf<<<(N_NODES*2 + T-1)/T, T, 0, stream>>>(mbuf, N_NODES*2);
  k_edge_e<<<(N_EDGES*2 + T-1)/T, T, 0, stream>>>(eidx, eattr, sd, ebuf, mbuf);
  k_fix_m<<<(N_NODES*2 + T-1)/T, T, 0, stream>>>(mbuf, den, N_NODES*2);
  k_edge_p<<<(N_EDGES*2 + T-1)/T, T, 0, stream>>>(eidx, ebuf, mbuf, den);
  k_fill_bias_rows<<<(N_NODES*HC + T-1)/T, T, 0, stream>>>(outb, b2, N_NODES, HC);
  k_edge_aggr<<<(int)(((size_t)N_EDGES*HC + T-1)/T), T, 0, stream>>>(eidx, ebuf, den, big, outb);

  // ---- pool + head ----
  k_pool<<<NB, HC, 0, stream>>>(outb, gp);
  k_fc1<<<(NB*HC + T-1)/T, T, 0, stream>>>(vf, gp, w_fc1, b_fc1, hc);
  k_fc2<<<1, 256, 0, stream>>>(hc, w_fc2, b_fc2, (float*)d_out);
}
